// TransformerBlock_89060441850473
// MI455X (gfx1250) — matrix operand order repaired and checked by reading, not compiled, not run on hardware
//
#include <hip/hip_runtime.h>

// ---------------------------------------------------------------------------
// Types for CDNA5 WMMA (wave32, 16x16x32 bf16 -> f32) and TDM descriptors
// ---------------------------------------------------------------------------
typedef __attribute__((ext_vector_type(16))) __bf16        v16bf;
typedef __attribute__((ext_vector_type(8)))  float         v8f;
typedef __attribute__((ext_vector_type(4)))  unsigned int  v4u;   // also TDM g0
typedef __attribute__((ext_vector_type(8)))  int           i32x8;
typedef __attribute__((ext_vector_type(4)))  int           i32x4;

union FragBF { v16bf v; v4u q[2]; };

#define DIM_    2048
#define NH_     16
#define HD_     128
#define SEQ_    2048
#define BATCH_  2
#define HIDDEN_ 5632

// --- TDM availability probing ----------------------------------------------
#if defined(__has_builtin)
#  if __has_builtin(__builtin_amdgcn_tensor_load_to_lds)
#    define HAVE_TDM 1
#  endif
#  if __has_builtin(__builtin_amdgcn_s_wait_tensorcnt)
#    define HAVE_TENSORCNT 1
#  endif
#endif
#ifndef HAVE_TDM
#  define HAVE_TDM 0
#endif
#ifndef HAVE_TENSORCNT
#  define HAVE_TENSORCNT 0
#endif
// therock-10.0 headers ship only with the 6-arg-builtin toolchain
#if defined(__has_include)
#  if __has_include(<hip/amd_detail/amd_gfx1250_TDM.h>)
#    define TDM_SIX_ARGS 1
#  endif
#endif
#ifndef TDM_SIX_ARGS
#  define TDM_SIX_ARGS 0
#endif

__device__ __forceinline__ unsigned short f2bf(float f) {
  unsigned int u = __builtin_bit_cast(unsigned int, f);
  u += 0x7FFFu + ((u >> 16) & 1u);           // round-to-nearest-even
  return (unsigned short)(u >> 16);
}

// Load a 16-element bf16 fragment as two 16-byte chunks.
__device__ __forceinline__ v16bf frag_ld(const unsigned short* p0,
                                         const unsigned short* p1) {
  FragBF f;
  f.q[0] = *(const v4u*)p0;
  f.q[1] = *(const v4u*)p1;
  return f.v;
}

__device__ __forceinline__ v8f wmma_bf16(v16bf a, v16bf b, v8f c) {
  return __builtin_amdgcn_wmma_f32_16x16x32_bf16(
      /*neg_a=*/false, a, /*neg_b=*/false, b,
      /*c_mod=*/(short)0, c, /*reuse_a=*/false, /*reuse_b=*/false);
}

__device__ __forceinline__ void tensorcnt_wait0() {
#if HAVE_TENSORCNT
  __builtin_amdgcn_s_wait_tensorcnt(0);
#else
  asm volatile("s_wait_tensorcnt 0" ::: "memory");
#endif
}

// Issue one TDM load: tile of 64 rows x 32 bf16 elems from W (row stride K
// elems) into LDS at lds_byte_addr (packed row-major, 64B per row).
// D# packing per CDNA5 ISA ch.8 (group0: count/lds/global/type, group1:
// data_size=2B, tensor_dim0=K, tensor_dim1=64, tile=32x64, dim0_stride=K).
__device__ __forceinline__ void tdm_load_tile_64x32(const unsigned short* gsrc,
                                                    unsigned lds_byte_addr,
                                                    int K) {
#if HAVE_TDM
  unsigned long long ga = (unsigned long long)(size_t)gsrc;
  v4u g0 = {0, 0, 0, 0};
  g0[0] = 1u;                                         // count=1 (user, load)
  g0[1] = lds_byte_addr;                              // LDS byte address
  g0[2] = (unsigned)ga;                               // global addr [31:0]
  g0[3] = (unsigned)((ga >> 32) & 0x01FFFFFFu)        // global addr [56:32]
          | 0x80000000u;                              // type=2 ("image")
  i32x8 g1 = {0, 0, 0, 0, 0, 0, 0, 0};
  g1[0] = 0x00010000;                                 // data_size=1 -> 2 bytes
  g1[1] = (int)(((unsigned)K & 0xFFFFu) << 16);       // tensor_dim0[15:0]
  g1[2] = (int)((((unsigned)K >> 16) & 0xFFFFu)       // tensor_dim0[31:16]
                | (64u << 16));                       // tensor_dim1 = 64 rows
  g1[3] = (int)(32u << 16);                           // tile_dim0 = 32 elems
  g1[4] = 64;                                         // tile_dim1 = 64 rows
  g1[5] = K;                                          // tensor_dim0_stride
  i32x4 gz = {0, 0, 0, 0};
#  if TDM_SIX_ARGS
  i32x8 z8 = {0, 0, 0, 0, 0, 0, 0, 0};
  __builtin_amdgcn_tensor_load_to_lds(g0, g1, gz, gz, z8, 0);
#  else
  __builtin_amdgcn_tensor_load_to_lds(g0, g1, gz, gz, 0);
#  endif
#else
  (void)gsrc; (void)lds_byte_addr; (void)K;
#endif
}

// ---------------------------------------------------------------------------
// f32 -> bf16 elementwise convert (weights)
// ---------------------------------------------------------------------------
__global__ void cvt_f32_bf16_kernel(const float* __restrict__ src,
                                    unsigned short* __restrict__ dst, long n) {
  long i = (long)blockIdx.x * blockDim.x + threadIdx.x;
  long stride = (long)gridDim.x * blockDim.x;
  for (; i < n; i += stride) dst[i] = f2bf(src[i]);
}

// ---------------------------------------------------------------------------
// RMSNorm: one block per row of D=2048, bf16 output
// ---------------------------------------------------------------------------
__global__ void __launch_bounds__(256)
rmsnorm_bf16_kernel(const float* __restrict__ x, const float* __restrict__ w,
                    unsigned short* __restrict__ out) {
  const int D = DIM_;
  long row = blockIdx.x;
  const float* xr = x + row * (long)D;
  float ss = 0.f;
  for (int i = threadIdx.x; i < D; i += 256) { float v = xr[i]; ss += v * v; }
#pragma unroll
  for (int off = 16; off > 0; off >>= 1) ss += __shfl_xor(ss, off, 32);
  __shared__ float red[8];
  if ((threadIdx.x & 31) == 0) red[threadIdx.x >> 5] = ss;
  __syncthreads();
  float tot = 0.f;
#pragma unroll
  for (int i = 0; i < 8; ++i) tot += red[i];
  float rinv = rsqrtf(tot / (float)D + 1e-5f);
  for (int i = threadIdx.x; i < D; i += 256)
    out[row * (long)D + i] = f2bf(xr[i] * rinv * w[i]);
}

// ---------------------------------------------------------------------------
// Main bf16 GEMM:  C[M,N](f32) = A[M,K](bf16) @ W[N,K]^T (bf16), optional
// fused residual.  One wave computes a 32(M) x 64(N) tile: 8 WMMAs per
// 12 b128 loads (B fragments shared across the two M-halves).
// ---------------------------------------------------------------------------
__global__ void __launch_bounds__(256)
gemm_bf16_kernel(const unsigned short* __restrict__ A,
                 const unsigned short* __restrict__ W,
                 const float* __restrict__ addsrc,
                 float* __restrict__ C, int M, int N, int K) {
  const int lane = threadIdx.x & 31;
  const int l16 = lane & 15;
  const int half = lane >> 4;
  long wid = (long)blockIdx.x * (blockDim.x >> 5) + (threadIdx.x >> 5);
  int tilesN = N >> 6;
  int m0 = (int)(wid / tilesN) << 5;
  int n0 = (int)(wid % tilesN) << 6;
  if (m0 >= M) return;

  const unsigned short* arow0 = A + (long)(m0 + l16) * K;
  const unsigned short* arow1 = arow0 + (long)16 * K;
  const unsigned short* wrow = W + (long)(n0 + l16) * K;
  const long nstride = (long)16 * K;

  const v8f vz = {0, 0, 0, 0, 0, 0, 0, 0};
  v8f accA[4], accB[4];
#pragma unroll
  for (int t = 0; t < 4; ++t) { accA[t] = vz; accB[t] = vz; }

  for (int k = 0; k < K; k += 32) {
    v16bf a0 = frag_ld(arow0 + k + half * 8, arow0 + k + 16 + half * 8);
    v16bf a1 = frag_ld(arow1 + k + half * 8, arow1 + k + 16 + half * 8);
    __builtin_prefetch(arow0 + k + 512, 0, 1);  // -> global_prefetch_b8
    __builtin_prefetch(arow1 + k + 512, 0, 1);
    int kb = k + half * 8;
#pragma unroll
    for (int t = 0; t < 4; ++t) {
      const unsigned short* wp = wrow + t * nstride + kb;
      v16bf b = frag_ld(wp, wp + 16);
      accA[t] = wmma_bf16(a0, b, accA[t]);
      accB[t] = wmma_bf16(a1, b, accB[t]);
    }
  }

#pragma unroll
  for (int r = 0; r < 8; ++r) {
    int rowA = m0 + r + half * 8;
    int rowB = rowA + 16;
    long baseA = (long)rowA * N + n0 + l16;
    long baseB = (long)rowB * N + n0 + l16;
#pragma unroll
    for (int t = 0; t < 4; ++t) {
      float va = accA[t][r], vb = accB[t][r];
      if (addsrc) {
        C[baseA + t * 16] = va + addsrc[baseA + t * 16];
        C[baseB + t * 16] = vb + addsrc[baseB + t * 16];
      } else {
        C[baseA + t * 16] = va;
        C[baseB + t * 16] = vb;
      }
    }
  }
}

// ---------------------------------------------------------------------------
// TDM-staged bf16 GEMM (used for the attention output projection):
// block = 8 waves sharing one 64-column B tile; wave0 streams 64x32 K-slices
// of W into double-buffered LDS with tensor_load_to_lds (TENSORcnt), the
// block barriers, and B fragments are read from LDS (ds_load_b128).
// Block tile: 128(M) x 64(N); wave tile 16 x 64.
// ---------------------------------------------------------------------------
__global__ void __launch_bounds__(256)
gemm_bf16_tdm_kernel(const unsigned short* __restrict__ A,
                     const unsigned short* __restrict__ W,
                     const float* __restrict__ addsrc,
                     float* __restrict__ C, int M, int N, int K) {
  __shared__ __align__(16) unsigned short Bt[2][64 * 32];
  const int lane = threadIdx.x & 31;
  const int l16 = lane & 15;
  const int half = lane >> 4;
  const int w = threadIdx.x >> 5;
  int tilesN = N >> 6;
  int mb = blockIdx.x / tilesN, nb = blockIdx.x % tilesN;
  int m0 = mb * 128 + w * 16;
  int n0 = nb * 64;

  const unsigned short* arow = A + (long)(m0 + l16) * K;
  const unsigned short* wbase = W + (long)n0 * K;

  const v8f vz = {0, 0, 0, 0, 0, 0, 0, 0};
  v8f acc[4];
#pragma unroll
  for (int t = 0; t < 4; ++t) acc[t] = vz;

#if HAVE_TDM
  // Preload first K-slice (wave0 drives the TDM).
  if (w == 0) {
    tdm_load_tile_64x32(wbase, (unsigned)(size_t)&Bt[0][0], K);
    tensorcnt_wait0();
  }
  __syncthreads();
#else
  // Fallback: cooperative staged copy (256 threads x 16B = 64x32 bf16 tile).
  {
    int row = threadIdx.x >> 2, ch = (threadIdx.x & 3) * 8;
    *(v4u*)&Bt[0][row * 32 + ch] = *(const v4u*)(wbase + (long)row * K + ch);
  }
  __syncthreads();
#endif

  int cur = 0;
  for (int k = 0; k < K; k += 32) {
    bool more = (k + 32) < K;
#if HAVE_TDM
    if (w == 0 && more)
      tdm_load_tile_64x32(wbase + k + 32,
                          (unsigned)(size_t)&Bt[cur ^ 1][0], K);
#else
    if (more) {
      int row = threadIdx.x >> 2, ch = (threadIdx.x & 3) * 8;
      *(v4u*)&Bt[cur ^ 1][row * 32 + ch] =
          *(const v4u*)(wbase + (long)row * K + k + 32 + ch);
    }
#endif
    v16bf a = frag_ld(arow + k + half * 8, arow + k + 16 + half * 8);
    __builtin_prefetch(arow + k + 512, 0, 1);
    const unsigned short* bl = &Bt[cur][l16 * 32 + half * 8];
#pragma unroll
    for (int t = 0; t < 4; ++t) {
      v16bf b = frag_ld(bl + t * 512, bl + t * 512 + 16);
      acc[t] = wmma_bf16(a, b, acc[t]);
    }
#if HAVE_TDM
    if (w == 0 && more) tensorcnt_wait0();
#endif
    __syncthreads();
    cur ^= 1;
  }

#pragma unroll
  for (int r = 0; r < 8; ++r) {
    int row = m0 + r + half * 8;
    long base = (long)row * N + n0 + l16;
#pragma unroll
    for (int t = 0; t < 4; ++t) {
      float v = acc[t][r];
      if (addsrc) C[base + t * 16] = v + addsrc[base + t * 16];
      else        C[base + t * 16] = v;
    }
  }
}

// ---------------------------------------------------------------------------
// RoPE + head split.  qkv f32 [B,S,3D] ->
//   Qb bf16 [B,H,S,HD], Kb bf16 [B,H,S,HD], Vt bf16 [B,H,HD,S] (transposed)
// ---------------------------------------------------------------------------
__global__ void __launch_bounds__(256)
rope_split_kernel(const float* __restrict__ qkv,
                  const float* __restrict__ fc, const float* __restrict__ fs,
                  unsigned short* __restrict__ Qb,
                  unsigned short* __restrict__ Kb,
                  unsigned short* __restrict__ Vt) {
  const int D = DIM_, S = SEQ_, H = NH_;
  long bs = blockIdx.x;
  int s = (int)(bs % S);
  int b = (int)(bs / S);
  const float* r = qkv + bs * (long)(3 * D);
  for (int p = threadIdx.x; p < D / 2; p += 256) {
    int h = p >> 6;        // HD/2 = 64 pairs per head
    int d2 = p & 63;
    float c = fc[s * 64 + d2], sn = fs[s * 64 + d2];
    int col = h * HD_ + 2 * d2;
    float qr = r[col], qi = r[col + 1];
    float kr = r[D + col], ki = r[D + col + 1];
    long qb = (((long)(b * H + h)) * S + s) * HD_ + 2 * d2;
    Qb[qb]     = f2bf(qr * c - qi * sn);
    Qb[qb + 1] = f2bf(qr * sn + qi * c);
    Kb[qb]     = f2bf(kr * c - ki * sn);
    Kb[qb + 1] = f2bf(kr * sn + ki * c);
    float v0 = r[2 * D + col], v1 = r[2 * D + col + 1];
    long vb = (((long)(b * H + h)) * HD_ + 2 * d2) * S + s;
    Vt[vb]     = f2bf(v0);
    Vt[vb + S] = f2bf(v1);
  }
}

// ---------------------------------------------------------------------------
// Flash attention (causal), one wave per (b,h, 16-query tile).
// ---------------------------------------------------------------------------
__global__ void __launch_bounds__(256)
flash_attn_kernel(const unsigned short* __restrict__ Qb,
                  const unsigned short* __restrict__ Kb,
                  const unsigned short* __restrict__ Vt,
                  unsigned short* __restrict__ O) {
  const int S = SEQ_, H = NH_, D = DIM_;
  __shared__ __align__(16) unsigned short plds[8 * 512];  // per-wave 16x32 P
  const int lane = threadIdx.x & 31;
  const int l16 = lane & 15, half = lane >> 4;
  const int widb = threadIdx.x >> 5;
  unsigned short* pl = plds + widb * 512;

  long gw = (long)blockIdx.x * 8 + widb;
  const int qtiles = S / 16;
  int bh = (int)(gw / qtiles);
  int qt = (int)(gw % qtiles);
  if (bh >= BATCH_ * H) return;
  int q0 = qt * 16;

  const unsigned short* Qp = Qb + (long)bh * S * HD_;
  const unsigned short* Kp = Kb + (long)bh * S * HD_;
  const unsigned short* Vp = Vt + (long)bh * HD_ * S;

  v16bf qf[4];
  const unsigned short* qrow = Qp + (long)(q0 + l16) * HD_;
#pragma unroll
  for (int c = 0; c < 4; ++c)
    qf[c] = frag_ld(qrow + c * 32 + half * 8, qrow + c * 32 + 16 + half * 8);

  v8f o[8];
  const v8f vz = {0, 0, 0, 0, 0, 0, 0, 0};
#pragma unroll
  for (int t = 0; t < 8; ++t) o[t] = vz;
  float mrun[8], lrun[8];
#pragma unroll
  for (int r = 0; r < 8; ++r) { mrun[r] = -3.0e38f; lrun[r] = 0.f; }

  const float scale = 0.088388347648318447f;  // 1/sqrt(128)
  int nkb = (q0 + 47) >> 5;                   // causal: keys <= q0+15
  for (int kb = 0; kb < nkb; ++kb) {
    int j0 = kb << 5;
    v8f s0 = vz, s1 = vz;
    const unsigned short* krow0 = Kp + (long)(j0 + l16) * HD_;
    const unsigned short* krow1 = krow0 + 16 * HD_;
#pragma unroll
    for (int c = 0; c < 4; ++c) {
      int kbase = c * 32 + half * 8;
      v16bf k0 = frag_ld(krow0 + kbase, krow0 + kbase + 16);
      v16bf k1 = frag_ld(krow1 + kbase, krow1 + kbase + 16);
      s0 = wmma_bf16(qf[c], k0, s0);
      s1 = wmma_bf16(qf[c], k1, s1);
    }
#pragma unroll
    for (int r = 0; r < 8; ++r) {
      int qrow_i = q0 + r + half * 8;
      float x0 = ((j0 + l16)      <= qrow_i) ? s0[r] * scale : -3.0e38f;
      float x1 = ((j0 + 16 + l16) <= qrow_i) ? s1[r] * scale : -3.0e38f;
      float rm = fmaxf(x0, x1);
#pragma unroll
      for (int off = 1; off < 16; off <<= 1) rm = fmaxf(rm, __shfl_xor(rm, off, 32));
      float mnew = fmaxf(mrun[r], rm);
      float alpha = __expf(mrun[r] - mnew);
      float p0 = __expf(x0 - mnew);
      float p1 = __expf(x1 - mnew);
      float rs = p0 + p1;
#pragma unroll
      for (int off = 1; off < 16; off <<= 1) rs += __shfl_xor(rs, off, 32);
      lrun[r] = lrun[r] * alpha + rs;
      mrun[r] = mnew;
#pragma unroll
      for (int t = 0; t < 8; ++t) o[t][r] *= alpha;
      int row = r + half * 8;
      pl[row * 32 + l16]      = f2bf(p0);
      pl[row * 32 + 16 + l16] = f2bf(p1);
    }
    asm volatile("s_wait_dscnt 0" ::: "memory");
    const unsigned short* prow = pl + l16 * 32 + half * 8;
    v16bf pf = frag_ld(prow, prow + 16);
#pragma unroll
    for (int t = 0; t < 8; ++t) {
      const unsigned short* vrow = Vp + (long)(t * 16 + l16) * S + j0 + half * 8;
      v16bf vf = frag_ld(vrow, vrow + 16);
      o[t] = wmma_bf16(pf, vf, o[t]);
    }
  }

  int h = bh % H, b = bh / H;
#pragma unroll
  for (int t = 0; t < 8; ++t) {
#pragma unroll
    for (int r = 0; r < 8; ++r) {
      int row = q0 + r + half * 8;
      long idx = ((long)b * S + row) * D + h * HD_ + t * 16 + l16;
      O[idx] = f2bf(o[t][r] / lrun[r]);
    }
  }
}

// ---------------------------------------------------------------------------
// SwiGLU gate: out_bf16 = silu(h1) * h3
// ---------------------------------------------------------------------------
__global__ void silu_mul_kernel(const float* __restrict__ h1,
                                const float* __restrict__ h3,
                                unsigned short* __restrict__ out, long n) {
  long i = (long)blockIdx.x * blockDim.x + threadIdx.x;
  long stride = (long)gridDim.x * blockDim.x;
  for (; i < n; i += stride) {
    float a = h1[i];
    float s = a / (1.f + __expf(-a));
    out[i] = f2bf(s * h3[i]);
  }
}

// ---------------------------------------------------------------------------
// Host launch
// ---------------------------------------------------------------------------
extern "C" void kernel_launch(void* const* d_in, const int* in_sizes, int n_in,
                              void* d_out, int out_size, void* d_ws, size_t ws_size,
                              hipStream_t stream) {
  const float* x     = (const float*)d_in[0];
  const float* fc    = (const float*)d_in[1];
  const float* fs    = (const float*)d_in[2];
  /* d_in[3] = mask: unused, causal mask applied analytically */
  const float* w_in  = (const float*)d_in[4];
  const float* w_out = (const float*)d_in[5];
  const float* w1    = (const float*)d_in[6];
  const float* w2    = (const float*)d_in[7];
  const float* w3    = (const float*)d_in[8];
  const float* anw   = (const float*)d_in[9];
  const float* fnw   = (const float*)d_in[10];

  const int B = BATCH_, S = SEQ_, D = DIM_, H = NH_, HID = HIDDEN_;
  const long BS = (long)B * S;

  char* p = (char*)d_ws;
  auto carve = [&](size_t bytes) -> void* {
    void* r = (void*)p;
    p += (bytes + 255) & ~(size_t)255;
    return r;
  };
  unsigned short* wInb  = (unsigned short*)carve((size_t)3 * D * D * 2);
  unsigned short* wOutb = (unsigned short*)carve((size_t)D * D * 2);
  unsigned short* w1b   = (unsigned short*)carve((size_t)HID * D * 2);
  unsigned short* w2b   = (unsigned short*)carve((size_t)D * HID * 2);
  unsigned short* w3b   = (unsigned short*)carve((size_t)HID * D * 2);
  unsigned short* hinb  = (unsigned short*)carve((size_t)BS * D * 2);
  float*          qkv   = (float*)carve((size_t)BS * 3 * D * 4);
  unsigned short* Qb    = (unsigned short*)carve((size_t)BS * D * 2);
  unsigned short* Kb    = (unsigned short*)carve((size_t)BS * D * 2);
  unsigned short* Vtb   = (unsigned short*)carve((size_t)BS * D * 2);
  unsigned short* attnb = (unsigned short*)carve((size_t)BS * D * 2);
  float*          hres  = (float*)carve((size_t)BS * D * 4);
  unsigned short* gb    = (unsigned short*)carve((size_t)BS * D * 2);
  float*          h1    = (float*)carve((size_t)BS * HID * 4);
  float*          h3    = (float*)carve((size_t)BS * HID * 4);
  unsigned short* hsb   = (unsigned short*)carve((size_t)BS * HID * 2);

  // 1) weights -> bf16
  cvt_f32_bf16_kernel<<<1024, 256, 0, stream>>>(w_in,  wInb,  (long)3 * D * D);
  cvt_f32_bf16_kernel<<<1024, 256, 0, stream>>>(w_out, wOutb, (long)D * D);
  cvt_f32_bf16_kernel<<<1024, 256, 0, stream>>>(w1,    w1b,   (long)HID * D);
  cvt_f32_bf16_kernel<<<1024, 256, 0, stream>>>(w2,    w2b,   (long)D * HID);
  cvt_f32_bf16_kernel<<<1024, 256, 0, stream>>>(w3,    w3b,   (long)HID * D);

  auto gemm = [&](const unsigned short* A, const unsigned short* W,
                  const float* add, float* C, int M, int N, int K) {
    long tiles = ((long)M / 32) * (N / 64);
    int blocks = (int)((tiles + 7) / 8);
    gemm_bf16_kernel<<<blocks, 256, 0, stream>>>(A, W, add, C, M, N, K);
  };

  // 2) attn RMSNorm
  rmsnorm_bf16_kernel<<<(int)BS, 256, 0, stream>>>(x, anw, hinb);
  // 3) QKV projection
  gemm(hinb, wInb, nullptr, qkv, (int)BS, 3 * D, D);
  // 4) RoPE + head split + V transpose
  rope_split_kernel<<<(int)BS, 256, 0, stream>>>(qkv, fc, fs, Qb, Kb, Vtb);
  // 5) causal flash attention
  {
    int waves = B * H * (S / 16);
    flash_attn_kernel<<<waves / 8, 256, 0, stream>>>(Qb, Kb, Vtb, attnb);
  }
  // 6) output projection + residual via TDM-staged GEMM: h = x + attn @ Wout^T
  {
    int blocks = ((int)BS / 128) * (D / 64);
    gemm_bf16_tdm_kernel<<<blocks, 256, 0, stream>>>(attnb, wOutb, x, hres,
                                                     (int)BS, D, D);
  }
  // 7) ffn RMSNorm
  rmsnorm_bf16_kernel<<<(int)BS, 256, 0, stream>>>(hres, fnw, gb);
  // 8-9) gate / up projections
  gemm(gb, w1b, nullptr, h1, (int)BS, HID, D);
  gemm(gb, w3b, nullptr, h3, (int)BS, HID, D);
  // 10) silu(h1)*h3 -> bf16
  silu_mul_kernel<<<2048, 256, 0, stream>>>(h1, h3, hsb, BS * (long)HID);
  // 11) down projection + residual -> d_out
  gemm(hsb, w2b, hres, (float*)d_out, (int)BS, D, HID);

  (void)in_sizes; (void)n_in; (void)out_size; (void)ws_size;
}
